// MultiHeadAttention_58626303590450
// MI455X (gfx1250) — matrix operand order repaired and checked by reading, not compiled, not run on hardware
//
#include <hip/hip_runtime.h>
#include <hip/hip_bf16.h>

typedef _Float16 v4h  __attribute__((ext_vector_type(4)));
typedef _Float16 v8h  __attribute__((ext_vector_type(8)));
typedef _Float16 v16h __attribute__((ext_vector_type(16)));
typedef float    v8f  __attribute__((ext_vector_type(8)));

#define SEQ    2048
#define EMBED  2048
#define NHEAD  32
#define HDIM   64
#define QKVD   6144
#define RSCALE 0.125f   // 64^-0.5

static __device__ __forceinline__ v16h cat8(v8h lo, v8h hi8) {
  return __builtin_shufflevector(lo, hi8, 0,1,2,3,4,5,6,7,8,9,10,11,12,13,14,15);
}
static __device__ __forceinline__ v8f wmma16(v16h a, v16h b, v8f c) {
  return __builtin_amdgcn_wmma_f32_16x16x32_f16(false, a, false, b, (short)0, c, false, false);
}
// CDNA5 async copy global->LDS (ASYNCcnt), 16B per lane.
static __device__ __forceinline__ void async_b128(_Float16* lds_dst, const _Float16* gsrc) {
  unsigned int loff = (unsigned int)(unsigned long long)lds_dst; // LDS aperture: addr[31:0] = LDS offset
  asm volatile("global_load_async_to_lds_b128 %0, %1, off" :: "v"(loff), "v"(gsrc) : "memory");
}
static __device__ __forceinline__ void wait_async0() {
  asm volatile("s_wait_asynccnt 0x0" ::: "memory");
}

// ---------------------------------------------------------------------------
// fp32 -> f16 elementwise conversion, 4 elements/thread
// ---------------------------------------------------------------------------
__global__ __launch_bounds__(256) void mha_cvt_f16(const float4* __restrict__ src,
                                                   _Float16* __restrict__ dst) {
  int t = blockIdx.x * 256 + threadIdx.x;
  float4 v = src[t];
  v4h o;
  o[0] = (_Float16)v.x; o[1] = (_Float16)v.y;
  o[2] = (_Float16)v.z; o[3] = (_Float16)v.w;
  *(v4h*)(dst + (size_t)t * 4) = o;
}

// ---------------------------------------------------------------------------
// C(MxN,f32) = A(MxK,f16 row-major) * B(NxK,f16 row-major)^T
// 8 waves / block; each wave computes a 32x64 tile (2 A frags x 4 B frags =
// 8 v_wmma_f32_16x16x32_f16 per 32-K chunk; each B frag reused twice).
// Fragment layouts per CDNA5 ISA 7.12.2:
//   A 16x32: lane<16 holds row M=lane, K{0-7,16-23}; lane>=16 K{8-15,24-31}
//   B 32x16: lane<16 holds col N=lane, K{0-7,16-23}; lane>=16 col lane-16, K{8-15,24-31}
// ---------------------------------------------------------------------------
__global__ __launch_bounds__(256) void mha_gemm_nt(const _Float16* __restrict__ A,
                                                   const _Float16* __restrict__ B,
                                                   float* __restrict__ C,
                                                   int M, int N, int K) {
  const int lane  = threadIdx.x & 31;
  const int wave  = threadIdx.x >> 5;
  const int row16 = lane & 15;
  const int hi    = lane >> 4;
  const int tilesN = N >> 6;
  const int id = blockIdx.x * 8 + wave;
  const int m0 = (id / tilesN) << 5;   // 32 rows
  const int n0 = (id % tilesN) << 6;   // 64 cols

  v8f acc[2][4] = {};
  const _Float16* arow0 = A + (size_t)(m0 + row16) * K;
  const _Float16* arow1 = arow0 + (size_t)16 * K;

  for (int k0 = 0; k0 < K; k0 += 32) {
    __builtin_prefetch(arow0 + k0 + 2048, 0, 1);   // stream-ahead hint (global_prefetch_b8)
    v16h a0 = cat8(*(const v8h*)(arow0 + k0 + hi * 8),
                   *(const v8h*)(arow0 + k0 + 16 + hi * 8));
    v16h a1 = cat8(*(const v8h*)(arow1 + k0 + hi * 8),
                   *(const v8h*)(arow1 + k0 + 16 + hi * 8));
#pragma unroll
    for (int j = 0; j < 4; ++j) {
      const _Float16* brow = B + (size_t)(n0 + j * 16 + row16) * K + k0 + hi * 8;
      v16h bf = cat8(*(const v8h*)brow, *(const v8h*)(brow + 16));
      acc[0][j] = wmma16(a0, bf, acc[0][j]);
      acc[1][j] = wmma16(a1, bf, acc[1][j]);
    }
  }
  // C frag: lane<16 -> rows M=r cols=lane ; lane>=16 -> rows M=8+r cols=lane-16
#pragma unroll
  for (int half = 0; half < 2; ++half) {
#pragma unroll
    for (int j = 0; j < 4; ++j) {
      int col = n0 + j * 16 + row16;
#pragma unroll
      for (int r = 0; r < 8; ++r) {
        C[(size_t)(m0 + half * 16 + hi * 8 + r) * N + col] = acc[half][j][r];
      }
    }
  }
}

// ---------------------------------------------------------------------------
// RoPE + pack: qkv32[N][6144] (d = which*2048 + h*64 + e) ->
//   Q [H][N][64] f16 (pre-scaled by 64^-0.5), K [H][N][64] f16, Vt [H][64][N] f16
// One thread per (h, n, j), j in [0,32): handles rope pair (2j,2j+1).
// ---------------------------------------------------------------------------
__global__ __launch_bounds__(256) void mha_rope_pack(const float* __restrict__ qkv,
                                                     _Float16* __restrict__ Q,
                                                     _Float16* __restrict__ Kd,
                                                     _Float16* __restrict__ Vt) {
  int t = blockIdx.x * 256 + threadIdx.x;   // t < 32*2048*32 = 2^21
  int j = t & 31;
  int n = (t >> 5) & (SEQ - 1);
  int h = t >> 16;

  const float* base = qkv + (size_t)n * QKVD + h * HDIM;
  float q1 = base[2 * j],             q2 = base[2 * j + 1];
  float k1 = base[EMBED + 2 * j],     k2 = base[EMBED + 2 * j + 1];
  float v1 = base[2 * EMBED + 2 * j], v2 = base[2 * EMBED + 2 * j + 1];

  // inv_freq = 10000^(-2j/64) ; log2(10000) = 13.2877123795...
  float invf = exp2f(-(float)(2 * j) * (13.287712379549449f / 64.0f));
  float ang  = (float)n * invf;
  float sn, cs;
  sincosf(ang, &sn, &cs);

  size_t qko = ((size_t)h * SEQ + n) * HDIM;
  Q[qko + j]       = (_Float16)((q1 * cs - q2 * sn) * RSCALE);
  Q[qko + 32 + j]  = (_Float16)((q2 * cs + q1 * sn) * RSCALE);
  Kd[qko + j]      = (_Float16)(k1 * cs - k2 * sn);
  Kd[qko + 32 + j] = (_Float16)(k2 * cs + k1 * sn);

  size_t vo = ((size_t)h * HDIM + 2 * j) * SEQ + n;
  Vt[vo]       = (_Float16)v1;
  Vt[vo + SEQ] = (_Float16)v2;
}

// ---------------------------------------------------------------------------
// Flash attention, one head per blockIdx.y, 8 waves x 16 queries per block.
// K/V chunks of 64 keys (16KB) double-buffered in LDS via async global->LDS
// copies; next chunk's copy overlaps the current chunk's 16 WMMAs + softmax.
// S^T = K_chunk * Q_tile^T so each lane-pair (n, n+16) owns all 16 keys of a
// group for one query -> online softmax with a single shfl_xor(16).
// ---------------------------------------------------------------------------
__global__ __launch_bounds__(256) void mha_flash_attn(const _Float16* __restrict__ Q,
                                                      const _Float16* __restrict__ Kd,
                                                      const _Float16* __restrict__ Vt,
                                                      _Float16* __restrict__ O) {
  __shared__ __align__(16) _Float16 sk[2][64 * HDIM];   // [buf][key_local][d]   8KB each
  __shared__ __align__(16) _Float16 sv[2][HDIM * 64];   // [buf][d][key_local]   8KB each

  const int lane  = threadIdx.x & 31;
  const int wave  = threadIdx.x >> 5;
  const int row16 = lane & 15;
  const int hi    = lane >> 4;
  const int h     = blockIdx.y;
  const int q0    = blockIdx.x * 128 + wave * 16;

  const _Float16* Qh = Q  + (size_t)h * SEQ * HDIM;
  const _Float16* Kh = Kd + (size_t)h * SEQ * HDIM;
  const _Float16* Vh = Vt + (size_t)h * HDIM * SEQ;

  const int t  = threadIdx.x;
  const int vd = t >> 2, vp = t & 3;

  // stage 64 keys (8KB K + 8KB Vt) into buffer `buf`, 4 async b128 per thread
  auto stage = [&](int buf, int c0) {
    const _Float16* ks = Kh + (size_t)c0 * HDIM + t * 16;
    _Float16* kl = &sk[buf][t * 16];
    async_b128(kl, ks);
    async_b128(kl + 8, ks + 8);
    const _Float16* vs = Vh + (size_t)vd * SEQ + c0 + vp * 16;
    _Float16* vl = &sv[buf][vd * 64 + vp * 16];
    async_b128(vl, vs);
    async_b128(vl + 8, vs + 8);
  };

  // ---- prologue: async-stage chunk 0 into buffer 0 ----
  stage(0, 0);

  // B frags of Q^T (d-chunk x query), kept resident: lane n<16 -> col n, K(d) 0-15
  v16h bq0 = *(const v16h*)(Qh + (size_t)(q0 + row16) * HDIM + 0  + hi * 16);
  v16h bq1 = *(const v16h*)(Qh + (size_t)(q0 + row16) * HDIM + 32 + hi * 16);

  float m_run = -__builtin_inff();
  float l_run = 0.0f;
  v8f oacc[4] = {};

  wait_async0();
  __syncthreads();

  for (int i = 0; i < SEQ / 64; ++i) {
    const int cur = i & 1;
    // ---- issue async copy of next chunk into the other buffer ----
    if (i + 1 < SEQ / 64) stage(cur ^ 1, (i + 1) * 64);

    const _Float16* skb = sk[cur];
    const _Float16* svb = sv[cur];

    // ---- S^T: 4 frags, group g = keys g*16 .. g*16+15 of the chunk ----
    v8f s[4] = {};
#pragma unroll
    for (int dblk = 0; dblk < 2; ++dblk) {
      int d0 = dblk * 32;
      v16h bq = dblk ? bq1 : bq0;
#pragma unroll
      for (int g = 0; g < 4; ++g) {
        const _Float16* kr = skb + (g * 16 + row16) * HDIM + d0 + hi * 16;
        v16h ak = *(const v16h*)kr;
        s[g] = wmma16(ak, bq, s[g]);
      }
    }

    // ---- online softmax (stats per query, replicated across lane pair) ----
    float cmax = -__builtin_inff();
#pragma unroll
    for (int g = 0; g < 4; ++g)
#pragma unroll
      for (int r = 0; r < 8; ++r) cmax = fmaxf(cmax, s[g][r]);
    cmax = fmaxf(cmax, __shfl_xor(cmax, 16, 32));
    float m_new = fmaxf(m_run, cmax);
    float alpha = __expf(m_run - m_new);

    float p[4][8];
    float lsum = 0.0f;
#pragma unroll
    for (int g = 0; g < 4; ++g)
#pragma unroll
      for (int r = 0; r < 8; ++r) {
        p[g][r] = __expf(s[g][r] - m_new);
        lsum += p[g][r];
      }
    lsum += __shfl_xor(lsum, 16, 32);
    l_run = l_run * alpha + lsum;
    m_run = m_new;
#pragma unroll
    for (int f = 0; f < 4; ++f) oacc[f] *= alpha;

    // ---- pack P into two B fragments: bp[0]=keys 0-31, bp[1]=keys 32-63 ----
    // lane<16 needs keys 0-7,16-23 of its query: own rows of pa and pb
    // lane>=16 needs keys 8-15,24-31: own rows of pa and pb
    v16h bp[2];
#pragma unroll
    for (int pr = 0; pr < 2; ++pr) {
      float* pa = p[2 * pr];       // keys pr*32 +  0..15
      float* pb = p[2 * pr + 1];   // keys pr*32 + 16..31
#pragma unroll
      for (int r = 0; r < 8; ++r) {
        float send = hi ? pa[r] : pb[r];
        float recv = __shfl_xor(send, 16, 32);
        bp[pr][r]     = (_Float16)pa[r];
        bp[pr][r + 8] = (_Float16)pb[r];
      }
    }

    // ---- O^T += Vt_tile(16x32) * P(32x16), 4 d-frags x 2 key-halves ----
#pragma unroll
    for (int f = 0; f < 4; ++f) {
      const _Float16* svr = svb + (f * 16 + row16) * 64;
      v16h av0 = cat8(*(const v8h*)(svr + hi * 8),
                      *(const v8h*)(svr + 16 + hi * 8));
      oacc[f] = wmma16(av0, bp[0], oacc[f]);
      v16h av1 = cat8(*(const v8h*)(svr + 32 + hi * 8),
                      *(const v8h*)(svr + 48 + hi * 8));
      oacc[f] = wmma16(av1, bp[1], oacc[f]);
    }

    // ---- wait for next chunk's async copy; release this buffer ----
    wait_async0();
    __syncthreads();
  }

  // ---- normalize and store: O^T frag f: d = f*16 + hi*8 + r, query = q0+row16
  float invl = 1.0f / l_run;
#pragma unroll
  for (int f = 0; f < 4; ++f) {
#pragma unroll
    for (int r = 0; r < 8; ++r) {
      O[(size_t)(q0 + row16) * EMBED + h * HDIM + f * 16 + hi * 8 + r] =
          (_Float16)(oacc[f][r] * invl);
    }
  }
}

// ---------------------------------------------------------------------------
// launch
// ---------------------------------------------------------------------------
extern "C" void kernel_launch(void* const* d_in, const int* in_sizes, int n_in,
                              void* d_out, int out_size, void* d_ws, size_t ws_size,
                              hipStream_t stream) {
  const float* x      = (const float*)d_in[0];
  const float* w_qkv  = (const float*)d_in[1];
  const float* w_proj = (const float*)d_in[2];
  float* out = (float*)d_out;

  char* ws = (char*)d_ws;
  const size_t MB = 1024 * 1024;
  _Float16* xh     = (_Float16*)(ws);                 //  8 MB
  _Float16* wqkvh  = (_Float16*)(ws + 8 * MB);        // 24 MB
  _Float16* wprojh = (_Float16*)(ws + 32 * MB);       //  8 MB
  float*    qkv32  = (float*)   (ws + 40 * MB);       // 48 MB
  _Float16* qh     = (_Float16*)(ws + 88 * MB);       //  8 MB
  _Float16* kh     = (_Float16*)(ws + 96 * MB);       //  8 MB
  _Float16* vth    = (_Float16*)(ws + 104 * MB);      //  8 MB
  _Float16* oh     = (_Float16*)(ws + 112 * MB);      //  8 MB  (total 120 MB)

  // 1) f32 -> f16 conversions
  mha_cvt_f16<<<(SEQ * EMBED) / 1024, 256, 0, stream>>>((const float4*)x, xh);
  mha_cvt_f16<<<(QKVD * EMBED) / 1024, 256, 0, stream>>>((const float4*)w_qkv, wqkvh);
  mha_cvt_f16<<<(EMBED * EMBED) / 1024, 256, 0, stream>>>((const float4*)w_proj, wprojh);

  // 2) qkv = x @ w_qkv^T   (2048 x 6144), 32x64 tiles
  {
    int tiles = (SEQ / 32) * (QKVD / 64);     // 6144 waves
    mha_gemm_nt<<<tiles / 8, 256, 0, stream>>>(xh, wqkvh, qkv32, SEQ, QKVD, EMBED);
  }

  // 3) RoPE + pack q/k/vT per head
  mha_rope_pack<<<(NHEAD * SEQ * 32) / 256, 256, 0, stream>>>(qkv32, qh, kh, vth);

  // 4) flash attention -> oh [N][C] f16
  {
    dim3 grid(SEQ / 128, NHEAD);
    mha_flash_attn<<<grid, 256, 0, stream>>>(qh, kh, vth, oh);
  }

  // 5) out = oh @ w_proj^T  (2048 x 2048, fp32 out)
  {
    int tiles = (SEQ / 32) * (EMBED / 64);    // 2048 waves
    mha_gemm_nt<<<tiles / 8, 256, 0, stream>>>(oh, wprojh, out, SEQ, EMBED, EMBED);
  }
}
